// Fusion_swin_transformer_net_9869834847048
// MI455X (gfx1250) — compile-verified, matrix operand order checked by reading, not run on hardware
//
#include <hip/hip_runtime.h>
#include <hip/hip_bf16.h>
#include <math.h>

// ---------------------------------------------------------------------------
// Types for WMMA (gfx1250, wave32)
// ---------------------------------------------------------------------------
typedef __attribute__((ext_vector_type(16))) _Float16 v16h;
typedef __attribute__((ext_vector_type(8)))  _Float16 v8h;
typedef __attribute__((ext_vector_type(8)))  float    v8f;

#define BATCH 8
#define HP 56
#define LTOK 3136              // 56*56
#define TOK 25088              // BATCH*LTOK
#define CDIM 128
#define NH 8
#define HD 16
#define WS 7
#define NWIN 512               // BATCH*64
#define SHIFT 3
#define SCALE_ATTN 0.25f       // 16^-0.5
#define HEADROWS 401408        // BATCH*50176

// ---------------------------------------------------------------------------
// WMMA helpers
// ---------------------------------------------------------------------------
// 16-bit A/B fragment, 16x32 tile, row-major source with leading dim `ld`.
// Lane l holds row m=l%16.  l<16: K = {0..7} U {16..23}; l>=16: +8.
__device__ inline v16h load_frag(const _Float16* __restrict__ base, int ld, int k0, int lane) {
  int m  = lane & 15;
  int kh = (lane & 16) ? 8 : 0;
  const _Float16* p = base + (size_t)m * ld + k0 + kh;
  v8h lo = *(const v8h*)p;        // K = kh .. kh+7
  v8h hi = *(const v8h*)(p + 16); // K = kh+16 .. kh+23
  v16h r;
#pragma unroll
  for (int e = 0; e < 8; ++e) { r[e] = lo[e]; r[e + 8] = hi[e]; }
  return r;
}

__device__ inline v8f wmma16(v16h a, v16h b, v8f c) {
  return __builtin_amdgcn_wmma_f32_16x16x32_f16(false, a, false, b, (short)0, c, false, false);
}

__device__ inline float gelu_f(float x) {
  const float c0 = 0.7978845608028654f, c1 = 0.044715f;
  return 0.5f * x * (1.f + tanhf(c0 * (x + c1 * x * x * x)));
}

// ---------------------------------------------------------------------------
// Weight prep: fp32 (K,N) -> f16 (N,K)  (transposed so B-fragments load contig)
// ---------------------------------------------------------------------------
__global__ void convert_transpose_f16(const float* __restrict__ src, _Float16* __restrict__ dst,
                                      int K, int N) {
  int idx = blockIdx.x * blockDim.x + threadIdx.x;
  if (idx >= K * N) return;
  int n = idx % N, k = idx / N;
  dst[(size_t)n * K + k] = (_Float16)src[idx];
}

// ---------------------------------------------------------------------------
// Patch embed: patchify + 16->128 matmul + LN(pn) + pos_embed  -> X fp32 (TOK,128)
// ---------------------------------------------------------------------------
__global__ __launch_bounds__(128)
void patch_embed_kernel(const float* __restrict__ vis, const float* __restrict__ pw,
                        const float* __restrict__ pb, const float* __restrict__ g,
                        const float* __restrict__ be, const float* __restrict__ pos,
                        float* __restrict__ X) {
  int t = blockIdx.x, c = threadIdx.x;
  int b = t / LTOK, l = t % LTOK;
  int py = l / HP, px = l % HP;
  __shared__ float patch[16];
  __shared__ float red[128];
  if (c < 16) {
    int r = c >> 2, cc = c & 3;
    patch[c] = vis[((size_t)b * 224 + py * 4 + r) * 224 + px * 4 + cc];
  }
  __syncthreads();
  float acc = pb[c];
#pragma unroll
  for (int k = 0; k < 16; ++k) acc += patch[k] * pw[k * CDIM + c];
  red[c] = acc; __syncthreads();
  for (int s = 64; s > 0; s >>= 1) { if (c < s) red[c] += red[c + s]; __syncthreads(); }
  float mu = red[0] * (1.f / CDIM); __syncthreads();
  float d = acc - mu;
  red[c] = d * d; __syncthreads();
  for (int s = 64; s > 0; s >>= 1) { if (c < s) red[c] += red[c + s]; __syncthreads(); }
  float var = red[0] * (1.f / CDIM);
  float y = d * rsqrtf(var + 1e-5f) * g[c] + be[c] + pos[(size_t)l * CDIM + c];
  X[(size_t)t * CDIM + c] = y;
}

// ---------------------------------------------------------------------------
// LN + (optional shift roll) + window partition -> f16 (TOK,128) in window order
// ---------------------------------------------------------------------------
__global__ __launch_bounds__(128)
void ln_window_kernel(const float* __restrict__ X, const float* __restrict__ g,
                      const float* __restrict__ be, _Float16* __restrict__ Ah, int shift) {
  int t = blockIdx.x, c = threadIdx.x;
  __shared__ float red[128];
  float a = X[(size_t)t * CDIM + c];
  red[c] = a; __syncthreads();
  for (int s = 64; s > 0; s >>= 1) { if (c < s) red[c] += red[c + s]; __syncthreads(); }
  float mu = red[0] * (1.f / CDIM); __syncthreads();
  float d = a - mu;
  red[c] = d * d; __syncthreads();
  for (int s = 64; s > 0; s >>= 1) { if (c < s) red[c] += red[c + s]; __syncthreads(); }
  float var = red[0] * (1.f / CDIM);
  float y = d * rsqrtf(var + 1e-5f) * g[c] + be[c];
  int b = t / LTOK, l = t % LTOK;
  int yy = l / HP, xx = l % HP;
  int ys = shift ? (yy + HP - SHIFT) % HP : yy;
  int xs = shift ? (xx + HP - SHIFT) % HP : xx;
  int wy = ys / WS, wx = xs / WS, iy = ys % WS, ix = xs % WS;
  size_t row = ((size_t)b * 64 + wy * 8 + wx) * 49 + iy * WS + ix;
  Ah[row * CDIM + c] = (_Float16)y;
}

// ---------------------------------------------------------------------------
// Plain LN -> f16 (TOK,128), optional ReLU (for norm2 and final normf)
// ---------------------------------------------------------------------------
__global__ __launch_bounds__(128)
void ln_kernel(const float* __restrict__ X, const float* __restrict__ g,
               const float* __restrict__ be, _Float16* __restrict__ out, int relu) {
  int t = blockIdx.x, c = threadIdx.x;
  __shared__ float red[128];
  float a = X[(size_t)t * CDIM + c];
  red[c] = a; __syncthreads();
  for (int s = 64; s > 0; s >>= 1) { if (c < s) red[c] += red[c + s]; __syncthreads(); }
  float mu = red[0] * (1.f / CDIM); __syncthreads();
  float d = a - mu;
  red[c] = d * d; __syncthreads();
  for (int s = 64; s > 0; s >>= 1) { if (c < s) red[c] += red[c + s]; __syncthreads(); }
  float var = red[0] * (1.f / CDIM);
  float y = d * rsqrtf(var + 1e-5f) * g[c] + be[c];
  if (relu) y = fmaxf(y, 0.f);
  out[(size_t)t * CDIM + c] = (_Float16)y;
}

// ---------------------------------------------------------------------------
// Generic WMMA GEMM: D(M,N) = act(A(M,K) @ Wt(N,K)^T + bias) [+resid]
// one wave per 16x16 tile; A,Wt f16 row-major; output fp32 or f16
// act: 0 = none, 1 = GELU
// ---------------------------------------------------------------------------
__global__ __launch_bounds__(32)
void gemm_f16_kernel(const _Float16* __restrict__ A, const _Float16* __restrict__ Wt,
                     const float* __restrict__ bias, const float* __restrict__ resid,
                     float* __restrict__ Cf, _Float16* __restrict__ Ch,
                     int N, int K, int act) {
  int mt = blockIdx.x, nt = blockIdx.y;
  int lane = threadIdx.x;
  size_t row0 = (size_t)mt * 16;
  int n0 = nt * 16;
  const _Float16* Abase = A + row0 * K;
  const _Float16* Bbase = Wt + (size_t)n0 * K;
  v8f acc = {};
  for (int k0 = 0; k0 < K; k0 += 32) {
    v16h af = load_frag(Abase, K, k0, lane);
    v16h bf = load_frag(Bbase, K, k0, lane);
    acc = wmma16(af, bf, acc);
  }
  int col = n0 + (lane & 15);
  size_t rbase = row0 + (size_t)((lane >> 4) << 3);
  float bv = bias ? bias[col] : 0.f;
#pragma unroll
  for (int j = 0; j < 8; ++j) {
    size_t r = rbase + j;
    float v = acc[j] + bv;
    if (act == 1) v = gelu_f(v);
    if (resid) v += resid[r * N + col];
    if (Ch) Ch[r * N + col] = (_Float16)v;
    else    Cf[r * N + col] = v;
  }
}

// ---------------------------------------------------------------------------
// Window attention core: per (window, head) 49x49 softmax attention
// QKV fp32 (TOK,384) in window-row order -> Oh f16 (TOK,128) window-row order
// ---------------------------------------------------------------------------
__global__ __launch_bounds__(64)
void attn_kernel(const float* __restrict__ QKV, const float* __restrict__ rpb_i,
                 _Float16* __restrict__ Oh, int shift) {
  int w = blockIdx.x;   // 0..511
  int h = blockIdx.y;   // 0..7
  int tid = threadIdx.x;
  __shared__ float q[49][16], k[49][16], v[49][16];
  for (int idx = tid; idx < 49 * 16; idx += 64) {
    int m = idx >> 4, d = idx & 15;
    const float* base = QKV + ((size_t)w * 49 + m) * 384 + h * 16 + d;
    q[m][d] = base[0] * SCALE_ATTN;
    k[m][d] = base[128];
    v[m][d] = base[256];
  }
  __syncthreads();
  if (tid < 49) {
    int n = tid;
    int ny = n / WS, nx = n % WS;
    int wi = w & 63, wy = wi >> 3, wx = wi & 7;
    int yn = wy * WS + ny, xn = wx * WS + nx;
    int cn = (yn < 49 ? 0 : (yn < 53 ? 1 : 2)) * 3 + (xn < 49 ? 0 : (xn < 53 ? 1 : 2));
    float s[49];
    float mx = -1e30f;
    for (int m = 0; m < 49; ++m) {
      float dot = 0.f;
#pragma unroll
      for (int dd = 0; dd < 16; ++dd) dot += q[n][dd] * k[m][dd];
      int my = m / WS, mxx = m % WS;
      dot += rpb_i[((ny - my + 6) * 13 + (nx - mxx + 6)) * NH + h];
      if (shift) {
        int ym = wy * WS + my, xm = wx * WS + mxx;
        int cm = (ym < 49 ? 0 : (ym < 53 ? 1 : 2)) * 3 + (xm < 49 ? 0 : (xm < 53 ? 1 : 2));
        if (cm != cn) dot -= 100.f;
      }
      s[m] = dot;
      mx = fmaxf(mx, dot);
    }
    float sum = 0.f;
    for (int m = 0; m < 49; ++m) { s[m] = __expf(s[m] - mx); sum += s[m]; }
    float inv = 1.f / sum;
#pragma unroll
    for (int dd = 0; dd < 16; ++dd) {
      float o = 0.f;
      for (int m = 0; m < 49; ++m) o += s[m] * v[m][dd];
      Oh[((size_t)w * 49 + n) * CDIM + h * 16 + dd] = (_Float16)(o * inv);
    }
  }
}

// ---------------------------------------------------------------------------
// Un-window + un-shift scatter + residual add:  Xout[t] = Xin[t] + P[window_row]
// ---------------------------------------------------------------------------
__global__ __launch_bounds__(256)
void unwindow_residual_kernel(const float* __restrict__ P, const float* __restrict__ Xin,
                              float* __restrict__ Xout, int shift) {
  int idx = blockIdx.x * 256 + threadIdx.x;
  if (idx >= TOK * CDIM) return;
  int c = idx & 127;
  int wr = idx >> 7;
  int iw = wr % 49, w = wr / 49;
  int b = w / 64, wi = w % 64;
  int wy = wi >> 3, wx = wi & 7, iy = iw / WS, ix = iw % WS;
  int ys = wy * WS + iy, xs = wx * WS + ix;
  int y = shift ? (ys + SHIFT) % HP : ys;
  int x = shift ? (xs + SHIFT) % HP : xs;
  size_t t = (size_t)b * LTOK + y * HP + x;
  Xout[t * CDIM + c] = Xin[t * CDIM + c] + P[(size_t)wr * CDIM + c];
}

// ---------------------------------------------------------------------------
// Head gather: replicate the raw-view reshape + 4x nearest upsample + raw view.
// F f16 (B,3136,128) -> G f16 (B*50176, 128)
// ---------------------------------------------------------------------------
__global__ __launch_bounds__(256)
void head_gather_kernel(const _Float16* __restrict__ F, _Float16* __restrict__ G) {
  size_t idx = (size_t)blockIdx.x * 256 + threadIdx.x;
  if (idx >= (size_t)HEADROWS * CDIM) return;
  int cc = (int)(idx & 127);
  size_t R = idx >> 7;               // b*50176 + r
  int b = (int)(R / 50176);
  int r = (int)(R % 50176);
  int ch = r / 392;                  // raw-view channel (50176 = 392*128)
  int p = (r % 392) * 128 + cc;      // flat within channel plane (224x224)
  int y2 = p / 224, x2 = p % 224;
  int yy = y2 >> 2, xx = x2 >> 2;    // nearest downmap to 56x56
  G[idx] = F[(size_t)b * (LTOK * CDIM) + ch * LTOK + yy * HP + xx];
}

// ---------------------------------------------------------------------------
// Fused head: out[r] = relu(G[r]@W1 + b1) . w2 + b2
// wave computes 16 rows x 256 cols; A frags loaded once, 16 N-tiles, in-register
// ReLU*w2 accumulate, cross-lane reduce.
// ---------------------------------------------------------------------------
__global__ __launch_bounds__(256)
void head_kernel(const _Float16* __restrict__ G, const _Float16* __restrict__ W1t,
                 const float* __restrict__ b1, const float* __restrict__ w2,
                 const float* __restrict__ b2, float* __restrict__ out) {
  int wave = threadIdx.x >> 5;
  int lane = threadIdx.x & 31;
  size_t mt = (size_t)blockIdx.x * 8 + wave;  // 25088 M-tiles total
  size_t row0 = mt * 16;
  const _Float16* Abase = G + row0 * CDIM;
  v16h a0 = load_frag(Abase, CDIM, 0,  lane);
  v16h a1 = load_frag(Abase, CDIM, 32, lane);
  v16h a2 = load_frag(Abase, CDIM, 64, lane);
  v16h a3 = load_frag(Abase, CDIM, 96, lane);
  float rsum[8];
#pragma unroll
  for (int j = 0; j < 8; ++j) rsum[j] = 0.f;
  float b2v = b2[0];
  for (int nt = 0; nt < 16; ++nt) {
    int n0 = nt * 16;
    const _Float16* Bbase = W1t + (size_t)n0 * CDIM;
    v8f acc = {};
    acc = wmma16(a0, load_frag(Bbase, CDIM, 0,  lane), acc);
    acc = wmma16(a1, load_frag(Bbase, CDIM, 32, lane), acc);
    acc = wmma16(a2, load_frag(Bbase, CDIM, 64, lane), acc);
    acc = wmma16(a3, load_frag(Bbase, CDIM, 96, lane), acc);
    int col = n0 + (lane & 15);
    float bb = b1[col], ww = w2[col];
#pragma unroll
    for (int j = 0; j < 8; ++j) rsum[j] += fmaxf(acc[j] + bb, 0.f) * ww;
  }
#pragma unroll
  for (int j = 0; j < 8; ++j) {
    float v = rsum[j];
    v += __shfl_xor(v, 1, 32);
    v += __shfl_xor(v, 2, 32);
    v += __shfl_xor(v, 4, 32);
    v += __shfl_xor(v, 8, 32);
    if ((lane & 15) == 0) {
      size_t r = row0 + (size_t)((lane >> 4) << 3) + j;
      out[r] = v + b2v;
    }
  }
}

// ---------------------------------------------------------------------------
// Host driver
// ---------------------------------------------------------------------------
extern "C" void kernel_launch(void* const* d_in, const int* in_sizes, int n_in,
                              void* d_out, int out_size, void* d_ws, size_t ws_size,
                              hipStream_t stream) {
  (void)in_sizes; (void)n_in; (void)out_size; (void)ws_size;
  const float* visible = (const float*)d_in[1];
  const float* patch_w = (const float*)d_in[2];
  const float* patch_b = (const float*)d_in[3];
  const float* pn_g    = (const float*)d_in[4];
  const float* pn_b    = (const float*)d_in[5];
  const float* pos     = (const float*)d_in[6];
  const float* norm1_g = (const float*)d_in[7];
  const float* norm1_b = (const float*)d_in[8];
  const float* qkv_w   = (const float*)d_in[9];
  const float* qkv_b   = (const float*)d_in[10];
  const float* proj_w  = (const float*)d_in[11];
  const float* proj_b  = (const float*)d_in[12];
  const float* norm2_g = (const float*)d_in[13];
  const float* norm2_b = (const float*)d_in[14];
  const float* fc1_w   = (const float*)d_in[15];
  const float* fc1_b   = (const float*)d_in[16];
  const float* fc2_w   = (const float*)d_in[17];
  const float* fc2_b   = (const float*)d_in[18];
  const float* rpb     = (const float*)d_in[19];
  const float* normf_g = (const float*)d_in[20];
  const float* normf_b = (const float*)d_in[21];
  const float* head1_w = (const float*)d_in[22];
  const float* head1_b = (const float*)d_in[23];
  const float* head2_w = (const float*)d_in[24];
  const float* head2_b = (const float*)d_in[25];
  float* out = (float*)d_out;

  // ---- workspace carve ----
  size_t off = 0;
  char* base = (char*)d_ws;
  auto carve = [&](size_t bytes) -> char* {
    char* p = base + off;
    off += (bytes + 255) & ~(size_t)255;
    return p;
  };
  _Float16* qkvT  = (_Float16*)carve((size_t)2 * 384 * 128 * 2);
  _Float16* projT = (_Float16*)carve((size_t)2 * 128 * 128 * 2);
  _Float16* fc1T  = (_Float16*)carve((size_t)2 * 512 * 128 * 2);
  _Float16* fc2T  = (_Float16*)carve((size_t)2 * 128 * 512 * 2);
  _Float16* h1T   = (_Float16*)carve((size_t)256 * 128 * 2);
  float*    Xa    = (float*)   carve((size_t)TOK * 128 * 4);
  float*    Xb    = (float*)   carve((size_t)TOK * 128 * 4);
  _Float16* Ah    = (_Float16*)carve((size_t)TOK * 128 * 2);   // also Mh / F
  _Float16* Oh    = (_Float16*)carve((size_t)TOK * 128 * 2);
  _Float16* Hh    = (_Float16*)carve((size_t)TOK * 512 * 2);
  float*    QKV   = (float*)   carve((size_t)TOK * 384 * 4);
  float*    P     = (float*)   carve((size_t)TOK * 128 * 4);
  _Float16* G     = (_Float16*)carve((size_t)HEADROWS * 128 * 2);

  // ---- weight prep (f16 transposed) ----
  auto cvt = [&](const float* s, _Float16* d, int K, int N) {
    int tot = K * N;
    convert_transpose_f16<<<(tot + 255) / 256, 256, 0, stream>>>(s, d, K, N);
  };
  for (int i = 0; i < 2; ++i) {
    cvt(qkv_w  + (size_t)i * 128 * 384, qkvT  + (size_t)i * 384 * 128, 128, 384);
    cvt(proj_w + (size_t)i * 128 * 128, projT + (size_t)i * 128 * 128, 128, 128);
    cvt(fc1_w  + (size_t)i * 128 * 512, fc1T  + (size_t)i * 512 * 128, 128, 512);
    cvt(fc2_w  + (size_t)i * 512 * 128, fc2T  + (size_t)i * 128 * 512, 512, 128);
  }
  cvt(head1_w, h1T, 128, 256);

  // ---- patch embed + LN + pos ----
  patch_embed_kernel<<<TOK, 128, 0, stream>>>(visible, patch_w, patch_b, pn_g, pn_b, pos, Xa);

  // ---- two swin blocks ----
  float* Xcur = Xa;   // input/output ping-pong: block leaves result in Xa
  for (int i = 0; i < 2; ++i) {
    int shift = (i == 1) ? 1 : 0;
    // LN1 + shift + window partition -> Ah (f16, window order)
    ln_window_kernel<<<TOK, 128, 0, stream>>>(Xcur, norm1_g + i * 128, norm1_b + i * 128, Ah, shift);
    // QKV GEMM (WMMA): (TOK,128)@(128,384)
    gemm_f16_kernel<<<dim3(TOK / 16, 384 / 16), 32, 0, stream>>>(
        Ah, qkvT + (size_t)i * 384 * 128, qkv_b + i * 384, nullptr, QKV, nullptr, 384, 128, 0);
    // attention core
    attn_kernel<<<dim3(NWIN, NH), 64, 0, stream>>>(QKV, rpb + (size_t)i * 169 * NH, Oh, shift);
    // proj GEMM (WMMA): (TOK,128)@(128,128) -> P fp32 (window order)
    gemm_f16_kernel<<<dim3(TOK / 16, 128 / 16), 32, 0, stream>>>(
        Oh, projT + (size_t)i * 128 * 128, proj_b + i * 128, nullptr, P, nullptr, 128, 128, 0);
    // un-window + un-shift + residual -> Xb
    unwindow_residual_kernel<<<(TOK * 128) / 256, 256, 0, stream>>>(P, Xcur, Xb, shift);
    // LN2 -> Ah (as Mh)
    ln_kernel<<<TOK, 128, 0, stream>>>(Xb, norm2_g + i * 128, norm2_b + i * 128, Ah, 0);
    // fc1 GEMM + GELU -> Hh f16
    gemm_f16_kernel<<<dim3(TOK / 16, 512 / 16), 32, 0, stream>>>(
        Ah, fc1T + (size_t)i * 512 * 128, fc1_b + i * 512, nullptr, nullptr, Hh, 512, 128, 1);
    // fc2 GEMM + residual(Xb) -> Xa
    gemm_f16_kernel<<<dim3(TOK / 16, 128 / 16), 32, 0, stream>>>(
        Hh, fc2T + (size_t)i * 128 * 512, fc2_b + i * 128, Xb, Xa, nullptr, 128, 512, 0);
    Xcur = Xa;
  }

  // ---- final LN + ReLU -> F (reuse Ah) ----
  ln_kernel<<<TOK, 128, 0, stream>>>(Xa, normf_g, normf_b, Ah, 1);

  // ---- raw-view + 4x nearest upsample + raw-view gather -> G ----
  {
    size_t tot = (size_t)HEADROWS * 128;
    head_gather_kernel<<<(unsigned)((tot + 255) / 256), 256, 0, stream>>>(Ah, G);
  }

  // ---- fused head: relu(G@W1+b1).w2 + b2 -> out (B*50176) ----
  head_kernel<<<HEADROWS / 16 / 8, 256, 0, stream>>>(G, h1T, head1_b, head2_w, head2_b, out);
}
